// PagedLLaMABlock_90074054131846
// MI455X (gfx1250) — compile-verified, matrix operand order checked by reading, not run on hardware
//
#include <hip/hip_runtime.h>
#include <hip/hip_bf16.h>
#include <math.h>

// ---------------------------------------------------------------------------
// LLaMA block forward: B=1, S=2048, D=4096, H=32, HD=128, FF=11008, f32 I/O.
// GEMMs + attention run on v_wmma_f32_16x16x32_bf16 (wave32 CDNA5 WMMA).
// ---------------------------------------------------------------------------

#define S_LEN 2048
#define D_DIM 4096
#define H_NUM 32
#define HD_DIM 128
#define FF_DIM 11008
#define EPSV 1e-5f

typedef __attribute__((ext_vector_type(16))) __bf16 v16bf;
typedef __attribute__((ext_vector_type(8)))  __bf16 v8bf;
typedef __attribute__((ext_vector_type(4)))  __bf16 v4bf;
typedef __attribute__((ext_vector_type(8)))  float  v8f;

static __device__ __forceinline__ v8f v8f_zero() {
    v8f z = {0.f, 0.f, 0.f, 0.f, 0.f, 0.f, 0.f, 0.f};
    return z;
}

static __device__ __forceinline__ v4bf cvt4(float4 f) {
    v4bf r = {(__bf16)f.x, (__bf16)f.y, (__bf16)f.z, (__bf16)f.w};
    return r;
}

// Load one 16x32 bf16 fragment (A-layout, also used for B via B^T tiles):
// lane&15 = row, lane>>4 selects K-half; K chunks of 8 live at p and p+16.
static __device__ __forceinline__ v16bf ld_frag(const __bf16* p) {
    union { v16bf v; v8bf h[2]; } u;
    u.h[0] = *reinterpret_cast<const v8bf*>(p);
    u.h[1] = *reinterpret_cast<const v8bf*>(p + 16);
    return u.v;
}

static __device__ __forceinline__ v8f wmma_bf16(v16bf a, v16bf b, v8f c) {
    return __builtin_amdgcn_wmma_f32_16x16x32_bf16(
        /*neg_a=*/false, a, /*neg_b=*/false, b,
        /*c_mod=*/(short)0, c, /*reuse_a=*/false, /*reuse_b=*/false);
}

// ---------------------------------------------------------------------------
// RMSNorm: one block per row (D=4096), 256 threads, float4 traffic.
// ---------------------------------------------------------------------------
__global__ __launch_bounds__(256)
void rmsnorm_kernel(const float* __restrict__ x, const float* __restrict__ w,
                    float* __restrict__ y) {
    __shared__ float red[256];
    const int tid = threadIdx.x;
    const size_t base = (size_t)blockIdx.x * D_DIM;

    float s = 0.f;
#pragma unroll
    for (int j = 0; j < 4; ++j) {
        const float4 f = *reinterpret_cast<const float4*>(x + base + j * 1024 + tid * 4);
        s += f.x * f.x + f.y * f.y + f.z * f.z + f.w * f.w;
    }
    red[tid] = s;
    __syncthreads();
#pragma unroll
    for (int off = 128; off > 0; off >>= 1) {
        if (tid < off) red[tid] += red[tid + off];
        __syncthreads();
    }
    const float inv = rsqrtf(red[0] * (1.0f / (float)D_DIM) + EPSV);

#pragma unroll
    for (int j = 0; j < 4; ++j) {
        const int idx = j * 1024 + tid * 4;
        float4 f = *reinterpret_cast<const float4*>(x + base + idx);
        const float4 wv = *reinterpret_cast<const float4*>(w + idx);
        f.x = f.x * inv * wv.x; f.y = f.y * inv * wv.y;
        f.z = f.z * inv * wv.z; f.w = f.w * inv * wv.w;
        *reinterpret_cast<float4*>(y + base + idx) = f;
    }
}

// ---------------------------------------------------------------------------
// GEMM: C[M,N] = A[M,K] @ W[K,N] (+ Res).  M=2048.  bf16 operands in LDS,
// f32 WMMA accumulate.  Block = 256 thr (8 waves), tile 128x128, BK=32.
// Wave (wm,wn) in 2x4 grid owns 64x32 -> 4x2 wmma accumulators.
// Register-staged double buffering: global b128 loads for slab k+1 issue
// before the WMMA batch of slab k.  EPI: 0 = plain, 1 = += Res.
// ---------------------------------------------------------------------------
template <int EPI>
__global__ __launch_bounds__(256)
void gemm_wmma(const float* __restrict__ A, const float* __restrict__ W,
               const float* __restrict__ Res, float* __restrict__ C,
               int K, int N) {
    // stride 48 bf16 = 96 B keeps every ds_load_b128 16-B aligned.
    __shared__ __bf16 As[128 * 48];
    __shared__ __bf16 Bs[128 * 48];

    const int tid  = threadIdx.x;
    const int lane = tid & 31;
    const int wave = tid >> 5;
    const int wm   = wave >> 2;      // 0..1  (M)
    const int wn   = wave & 3;       // 0..3  (N)
    const int row0 = blockIdx.y * 128;
    const int col0 = blockIdx.x * 128;
    const int r    = lane & 15;
    const int kb8  = (lane >> 4) * 8;

    // A staging: slot idx -> (row, col4).  B staging: slot idx -> (k, n4).
    const int aR[4] = { (tid + 0)   >> 3, (tid + 256) >> 3,
                        (tid + 512) >> 3, (tid + 768) >> 3 };
    const int aC   = (tid & 7) * 4;
    const int bK[4] = { (tid + 0)   >> 5, (tid + 256) >> 5,
                        (tid + 512) >> 5, (tid + 768) >> 5 };
    const int bN   = (tid & 31) * 4;

    v8f acc[4][2];
#pragma unroll
    for (int i = 0; i < 4; ++i)
#pragma unroll
        for (int j = 0; j < 2; ++j) acc[i][j] = v8f_zero();

    float4 regA[4], regB[4];
    // preload slab 0
#pragma unroll
    for (int j = 0; j < 4; ++j) {
        regA[j] = *reinterpret_cast<const float4*>(A + (size_t)(row0 + aR[j]) * K + aC);
        regB[j] = *reinterpret_cast<const float4*>(W + (size_t)bK[j] * N + col0 + bN);
    }

    for (int k0 = 0; k0 < K; k0 += 32) {
        // stage current slab regs -> LDS (bf16)
#pragma unroll
        for (int j = 0; j < 4; ++j) {
            *reinterpret_cast<v4bf*>(&As[aR[j] * 48 + aC]) = cvt4(regA[j]);
            const v4bf b = cvt4(regB[j]);
            Bs[(bN + 0) * 48 + bK[j]] = b[0];
            Bs[(bN + 1) * 48 + bK[j]] = b[1];
            Bs[(bN + 2) * 48 + bK[j]] = b[2];
            Bs[(bN + 3) * 48 + bK[j]] = b[3];
        }
        __syncthreads();

        // issue next slab's global loads before the WMMA batch
        const int k1 = k0 + 32;
        if (k1 < K) {
#pragma unroll
            for (int j = 0; j < 4; ++j) {
                regA[j] = *reinterpret_cast<const float4*>(
                    A + (size_t)(row0 + aR[j]) * K + k1 + aC);
                regB[j] = *reinterpret_cast<const float4*>(
                    W + (size_t)(k1 + bK[j]) * N + col0 + bN);
            }
            __builtin_prefetch(A + (size_t)(row0 + aR[0]) * K + k1 + 32 + aC, 0, 0);
            __builtin_prefetch(W + (size_t)(k1 + 32 + bK[0]) * N + col0 + bN, 0, 0);
        }

        // fragment loads (batched), then 8 back-to-back WMMAs
        v16bf af[4], bfr[2];
#pragma unroll
        for (int mt = 0; mt < 4; ++mt)
            af[mt] = ld_frag(&As[(wm * 64 + mt * 16 + r) * 48 + kb8]);
#pragma unroll
        for (int nt = 0; nt < 2; ++nt)
            bfr[nt] = ld_frag(&Bs[(wn * 32 + nt * 16 + r) * 48 + kb8]);

#pragma unroll
        for (int mt = 0; mt < 4; ++mt)
#pragma unroll
            for (int nt = 0; nt < 2; ++nt)
                acc[mt][nt] = wmma_bf16(af[mt], bfr[nt], acc[mt][nt]);
        __syncthreads();
    }

    // --- epilogue: C layout lane&15 = col, VGPR v -> row v + (lane>>4)*8 ---
#pragma unroll
    for (int mt = 0; mt < 4; ++mt) {
        const int mrow = row0 + wm * 64 + mt * 16 + (lane >> 4) * 8;
#pragma unroll
        for (int nt = 0; nt < 2; ++nt) {
            const int col = col0 + wn * 32 + nt * 16 + (lane & 15);
#pragma unroll
            for (int v = 0; v < 8; ++v) {
                const size_t off = (size_t)(mrow + v) * N + col;
                float val = acc[mt][nt][v];
                if (EPI == 1) val += Res[off];
                C[off] = val;
            }
        }
    }
}

// ---------------------------------------------------------------------------
// RoPE in place on q and k: grid (S, H), 64 threads = 64 (even,odd) pairs.
// ---------------------------------------------------------------------------
__global__ __launch_bounds__(64)
void rope_kernel(float* __restrict__ q, float* __restrict__ k) {
    const int s = blockIdx.x, h = blockIdx.y, i = threadIdx.x;
    const float inv_freq = __powf(10000.0f, -(float)i / 64.0f);
    float sn, cs;
    __sincosf((float)s * inv_freq, &sn, &cs);
    const size_t base = ((size_t)s * H_NUM + h) * HD_DIM;

    float x1 = q[base + 2 * i], x2 = q[base + 2 * i + 1];
    q[base + 2 * i]     = x1 * cs - x2 * sn;
    q[base + 2 * i + 1] = x1 * sn + x2 * cs;

    x1 = k[base + 2 * i]; x2 = k[base + 2 * i + 1];
    k[base + 2 * i]     = x1 * cs - x2 * sn;
    k[base + 2 * i + 1] = x1 * sn + x2 * cs;
}

// ---------------------------------------------------------------------------
// Causal flash attention.  Grid (S/64, H), 128 thr = 4 waves; each wave owns
// 16 query rows.  Per 32-key block: QK^T (8 wmma), online softmax with
// 16-lane shfl_xor row reductions, P->LDS->A-frag, P@V (8 wmma).
// Fragments are batch-preloaded so WMMAs issue back-to-back; causal masking
// is skipped (wave-uniform branch) for fully-unmasked key blocks.
// ---------------------------------------------------------------------------
__global__ __launch_bounds__(128)
void attn_kernel(const float* __restrict__ q, const float* __restrict__ k,
                 const float* __restrict__ v, float* __restrict__ o) {
    __shared__ __bf16 Ks[32 * 128];       // [key][d]   (= B^T of K^T)
    __shared__ __bf16 Vt[128 * 32];       // [d][key]   (= B^T of V)
    __shared__ __bf16 Ps[4 * 16 * 32];    // per-wave P tile

    const int tid  = threadIdx.x;
    const int lane = tid & 31;
    const int wv   = tid >> 5;
    const int h    = blockIdx.y;
    const int qtile = blockIdx.x;
    const int qbase = qtile * 64 + wv * 16;
    const int r    = lane & 15;
    const int half = lane >> 4;
    const int kb8  = half * 8;

    // Q fragments for this wave's 16 rows (K=128 -> 4 chunks of 32)
    v16bf qf[4];
    {
        const float* qp = q + ((size_t)(qbase + r) * H_NUM + h) * HD_DIM;
#pragma unroll
        for (int kt = 0; kt < 4; ++kt) {
            union { v16bf vv; __bf16 e[16]; } u;
#pragma unroll
            for (int j = 0; j < 8; ++j) u.e[j]     = (__bf16)qp[kt * 32 + kb8 + j];
#pragma unroll
            for (int j = 0; j < 8; ++j) u.e[8 + j] = (__bf16)qp[kt * 32 + kb8 + 16 + j];
            qf[kt] = u.vv;
        }
    }

    float mrow[8], lrow[8];
    v8f oacc[8];
#pragma unroll
    for (int i = 0; i < 8; ++i) { mrow[i] = -1e30f; lrow[i] = 0.f; oacc[i] = v8f_zero(); }

    const int nkb = 2 * (qtile + 1);          // causal key-block count
    for (int kb = 0; kb < nkb; ++kb) {
        // stage K tile (b64 stores) and transposed V tile (b16 scatter)
#pragma unroll
        for (int j = 0; j < 8; ++j) {
            const int idx = tid + j * 128;    // 0..1023 float4 slots
            const int key = idx >> 5;         // 0..31
            const int d4  = (idx & 31) * 4;   // 0..124
            const size_t g = ((size_t)(kb * 32 + key) * H_NUM + h) * HD_DIM + d4;
            *reinterpret_cast<v4bf*>(&Ks[key * 128 + d4]) =
                cvt4(*reinterpret_cast<const float4*>(k + g));
            const v4bf vb = cvt4(*reinterpret_cast<const float4*>(v + g));
            Vt[(d4 + 0) * 32 + key] = vb[0];
            Vt[(d4 + 1) * 32 + key] = vb[1];
            Vt[(d4 + 2) * 32 + key] = vb[2];
            Vt[(d4 + 3) * 32 + key] = vb[3];
        }
        __syncthreads();

        // S = Q K^T : preload all 8 K-frags, then 8 back-to-back WMMAs
        v16bf kf[2][4];
#pragma unroll
        for (int nt = 0; nt < 2; ++nt)
#pragma unroll
            for (int kt = 0; kt < 4; ++kt)
                kf[nt][kt] = ld_frag(&Ks[(nt * 16 + r) * 128 + kt * 32 + kb8]);

        v8f sacc[2] = {v8f_zero(), v8f_zero()};
#pragma unroll
        for (int nt = 0; nt < 2; ++nt)
#pragma unroll
            for (int kt = 0; kt < 4; ++kt)
                sacc[nt] = wmma_bf16(qf[kt], kf[nt][kt], sacc[nt]);

        // online softmax; skip masking for fully-unmasked tiles (wave-uniform)
        const float scale = 0.08838834764831845f;   // 1/sqrt(128)
        const bool diag = (kb * 32 + 31 > qbase);
#pragma unroll
        for (int vv = 0; vv < 8; ++vv) {
            float s0 = sacc[0][vv] * scale;
            float s1 = sacc[1][vv] * scale;
            if (diag) {
                const int qrow = qbase + half * 8 + vv;
                if (kb * 32 + r > qrow)      s0 = -1e30f;
                if (kb * 32 + 16 + r > qrow) s1 = -1e30f;
            }
            float t = fmaxf(s0, s1);
            t = fmaxf(t, __shfl_xor(t, 1, 16));
            t = fmaxf(t, __shfl_xor(t, 2, 16));
            t = fmaxf(t, __shfl_xor(t, 4, 16));
            t = fmaxf(t, __shfl_xor(t, 8, 16));
            const float mnew  = fmaxf(mrow[vv], t);
            const float alpha = __expf(mrow[vv] - mnew);
            const float p0 = __expf(s0 - mnew);
            const float p1 = __expf(s1 - mnew);
            float rs = p0 + p1;
            rs += __shfl_xor(rs, 1, 16);
            rs += __shfl_xor(rs, 2, 16);
            rs += __shfl_xor(rs, 4, 16);
            rs += __shfl_xor(rs, 8, 16);
            lrow[vv] = lrow[vv] * alpha + rs;
            mrow[vv] = mnew;
#pragma unroll
            for (int dt = 0; dt < 8; ++dt) oacc[dt][vv] *= alpha;
            const int pm = half * 8 + vv;
            Ps[(wv * 16 + pm) * 32 + r]      = (__bf16)p0;
            Ps[(wv * 16 + pm) * 32 + 16 + r] = (__bf16)p1;
        }

        // O += P @ V : preload P-frag + 8 V-frags, then 8 back-to-back WMMAs
        const v16bf pf = ld_frag(&Ps[(wv * 16 + r) * 32 + kb8]);
        v16bf vf[8];
#pragma unroll
        for (int dt = 0; dt < 8; ++dt)
            vf[dt] = ld_frag(&Vt[(dt * 16 + r) * 32 + kb8]);
#pragma unroll
        for (int dt = 0; dt < 8; ++dt)
            oacc[dt] = wmma_bf16(pf, vf[dt], oacc[dt]);
        __syncthreads();
    }

    // write normalized output: (s, h, hd) flattened == row-major (s, 4096)
#pragma unroll
    for (int dt = 0; dt < 8; ++dt)
#pragma unroll
        for (int vv = 0; vv < 8; ++vv) {
            const int srow = qbase + half * 8 + vv;
            const int d    = dt * 16 + r;
            o[(size_t)srow * D_DIM + h * HD_DIM + d] = oacc[dt][vv] / lrow[vv];
        }
}

// ---------------------------------------------------------------------------
// h = silu(g) * u, in place on g.  float4 traffic.
// ---------------------------------------------------------------------------
__global__ __launch_bounds__(256)
void silu_mul_kernel(float* __restrict__ g, const float* __restrict__ u) {
    const size_t i = ((size_t)blockIdx.x * blockDim.x + threadIdx.x) * 4;
    float4 gv = *reinterpret_cast<const float4*>(g + i);
    const float4 uv = *reinterpret_cast<const float4*>(u + i);
    gv.x = gv.x / (1.f + __expf(-gv.x)) * uv.x;
    gv.y = gv.y / (1.f + __expf(-gv.y)) * uv.y;
    gv.z = gv.z / (1.f + __expf(-gv.z)) * uv.z;
    gv.w = gv.w / (1.f + __expf(-gv.w)) * uv.w;
    *reinterpret_cast<float4*>(g + i) = gv;
}

// ---------------------------------------------------------------------------
// Launch
// ---------------------------------------------------------------------------
extern "C" void kernel_launch(void* const* d_in, const int* in_sizes, int n_in,
                              void* d_out, int out_size, void* d_ws, size_t ws_size,
                              hipStream_t stream) {
    const float* x     = (const float*)d_in[0];
    const float* ln_w  = (const float*)d_in[1];
    const float* ffl_w = (const float*)d_in[2];
    const float* wq    = (const float*)d_in[3];
    const float* wk    = (const float*)d_in[4];
    const float* wvw   = (const float*)d_in[5];
    const float* wo    = (const float*)d_in[6];
    const float* wg    = (const float*)d_in[7];
    const float* w1    = (const float*)d_in[8];
    const float* w2    = (const float*)d_in[9];
    float* out = (float*)d_out;
    float* ws  = (float*)d_ws;

    const size_t A = (size_t)S_LEN * D_DIM;    // 8,388,608
    const size_t G = (size_t)S_LEN * FF_DIM;   // 22,544,384
    float* xn   = ws;          // rmsnorm(x); later reused as attn output
    float* qb   = ws + A;      // q; later reused as xn2
    float* kb   = ws + 2 * A;
    float* vb   = ws + 3 * A;
    float* x2   = ws + 4 * A;
    float* gb   = ws + 5 * A;
    float* ub   = ws + 5 * A + G;
    float* attn = xn;
    float* xn2  = qb;

    const dim3 gemmD(D_DIM / 128, S_LEN / 128);   // (32, 16)
    const dim3 gemmF(FF_DIM / 128, S_LEN / 128);  // (86, 16)

    // 1) xn = rmsnorm(x, ln_w)
    rmsnorm_kernel<<<S_LEN, 256, 0, stream>>>(x, ln_w, xn);
    // 2) q/k/v projections
    gemm_wmma<0><<<gemmD, 256, 0, stream>>>(xn, wq,  nullptr, qb, D_DIM, D_DIM);
    gemm_wmma<0><<<gemmD, 256, 0, stream>>>(xn, wk,  nullptr, kb, D_DIM, D_DIM);
    gemm_wmma<0><<<gemmD, 256, 0, stream>>>(xn, wvw, nullptr, vb, D_DIM, D_DIM);
    // 3) RoPE on q, k
    rope_kernel<<<dim3(S_LEN, H_NUM), 64, 0, stream>>>(qb, kb);
    // 4) causal attention -> attn (reuses xn slot)
    attn_kernel<<<dim3(S_LEN / 64, H_NUM), 128, 0, stream>>>(qb, kb, vb, attn);
    // 5) x2 = attn @ wo + x
    gemm_wmma<1><<<gemmD, 256, 0, stream>>>(attn, wo, x, x2, D_DIM, D_DIM);
    // 6) xn2 = rmsnorm(x2, ff_ln_w)
    rmsnorm_kernel<<<S_LEN, 256, 0, stream>>>(x2, ffl_w, xn2);
    // 7) g = xn2 @ wg ; u = xn2 @ w1
    gemm_wmma<0><<<gemmF, 256, 0, stream>>>(xn2, wg, nullptr, gb, D_DIM, FF_DIM);
    gemm_wmma<0><<<gemmF, 256, 0, stream>>>(xn2, w1, nullptr, ub, D_DIM, FF_DIM);
    // 8) g = silu(g) * u
    silu_mul_kernel<<<(unsigned)(G / (256 * 4)), 256, 0, stream>>>(gb, ub);
    // 9) out = g @ w2 + x2
    gemm_wmma<1><<<gemmD, 256, 0, stream>>>(gb, w2, x2, out, FF_DIM, D_DIM);
}
